// QuantizedViTAttention_36266703848211
// MI455X (gfx1250) — 16-bit and 8-bit matrix operand order repaired and checked by reading, compiled, not run on hardware
//
#include <hip/hip_runtime.h>
#include <hip/hip_bf16.h>

constexpr int NHn = 16;
constexpr int HDn = 64;
constexpr int Hn  = 1024;
constexpr int Bn  = 8;
constexpr int Sn  = 577;
constexpr long Mn = (long)Bn * Sn;
constexpr int SPAD = 608;
constexpr int BH  = Bn * NHn;
constexpr float SCALING = 0.125f;

typedef __attribute__((ext_vector_type(8)))  int      v8i;
typedef __attribute__((ext_vector_type(8)))  float    v8f;
typedef __attribute__((ext_vector_type(16))) _Float16 v16h;
typedef __attribute__((ext_vector_type(8)))  _Float16 v8h;

constexpr size_t al256(size_t x) { return (x + 255) & ~(size_t)255; }
constexpr size_t OFF_SCAL = 0;
constexpr size_t OFF_X8   = 256;
constexpr size_t SZ_X8    = (size_t)Mn * Hn;
constexpr size_t OFF_W8   = al256(OFF_X8 + SZ_X8);
constexpr size_t SZ_W8    = 4ull * Hn * Hn;
constexpr size_t OFF_SW   = al256(OFF_W8 + SZ_W8);
constexpr size_t SZ_SW    = 4ull * Hn * sizeof(float);
constexpr size_t OFF_PROJ = al256(OFF_SW + SZ_SW);
constexpr size_t SZ_PROJ  = 3ull * Mn * Hn * sizeof(float);
constexpr size_t OFF_Q8   = al256(OFF_PROJ + SZ_PROJ);
constexpr size_t SZ_HEAD8 = (size_t)BH * Sn * HDn;
constexpr size_t OFF_K8   = al256(OFF_Q8 + SZ_HEAD8);
constexpr size_t OFF_VT   = al256(OFF_K8 + SZ_HEAD8);
constexpr size_t SZ_VT    = (size_t)BH * HDn * SPAD * sizeof(_Float16);
constexpr size_t OFF_ATTN = al256(OFF_VT + SZ_VT);
constexpr size_t SZ_ATTN  = (size_t)BH * Sn * Sn * sizeof(float);
constexpr size_t OFF_PRB  = al256(OFF_ATTN + SZ_ATTN);

__device__ __forceinline__ int iclamp_hi(int v, int hi) { return v < hi ? v : hi; }

__device__ __forceinline__ v8i load_a_i8(const signed char* rowk0, int grp) {
  int2 a0 = *(const int2*)(rowk0 +  0 + grp * 16);
  int2 a1 = *(const int2*)(rowk0 +  8 + grp * 16);
  int2 a2 = *(const int2*)(rowk0 + 32 + grp * 16);
  int2 a3 = *(const int2*)(rowk0 + 40 + grp * 16);
  v8i r; r[0]=a0.x; r[1]=a0.y; r[2]=a1.x; r[3]=a1.y; r[4]=a2.x; r[5]=a2.y; r[6]=a3.x; r[7]=a3.y;
  return r;
}
__device__ __forceinline__ v8i load_b_i8(const signed char* colk0, int grp) {
  int4 b0 = *(const int4*)(colk0 +  0 + grp * 16);
  int4 b1 = *(const int4*)(colk0 + 32 + grp * 16);
  v8i r; r[0]=b0.x; r[1]=b0.y; r[2]=b0.z; r[3]=b0.w; r[4]=b1.x; r[5]=b1.y; r[6]=b1.z; r[7]=b1.w;
  return r;
}

__global__ __launch_bounds__(256) void k_init(unsigned* scal, _Float16* vt, long nvt) {
  if (blockIdx.x == 0 && threadIdx.x < 16) scal[threadIdx.x] = 0u;
  for (long i = (long)blockIdx.x * 256 + threadIdx.x; i < nvt; i += (long)gridDim.x * 256)
    vt[i] = (_Float16)0.0f;
}

__global__ __launch_bounds__(256) void k_absmax(const float* __restrict__ x, long n, unsigned* slot) {
  float m = 0.0f;
  for (long i = (long)blockIdx.x * 256 + threadIdx.x; i < n; i += (long)gridDim.x * 256)
    m = fmaxf(m, fabsf(x[i]));
  __shared__ float red[256];
  red[threadIdx.x] = m; __syncthreads();
  for (int s = 128; s > 0; s >>= 1) {
    if (threadIdx.x < s) red[threadIdx.x] = fmaxf(red[threadIdx.x], red[threadIdx.x + s]);
    __syncthreads();
  }
  if (threadIdx.x == 0) atomicMax(slot, __float_as_uint(red[0]));
}

__global__ __launch_bounds__(256) void k_quant8(const float* __restrict__ x, const unsigned* slot,
                                                signed char* __restrict__ o, long n) {
  long i = (long)blockIdx.x * 256 + threadIdx.x;
  if (i >= n) return;
  float s = fmaxf(__uint_as_float(*slot), 1e-8f) * (1.0f / 127.0f);
  float q = fminf(fmaxf(roundf(x[i] / s), -128.0f), 127.0f);
  o[i] = (signed char)q;
}

__global__ __launch_bounds__(256) void k_quantw(const float* __restrict__ Wq, const float* __restrict__ Wk,
                                                const float* __restrict__ Wv, const float* __restrict__ Wo,
                                                signed char* __restrict__ w8, float* __restrict__ sw) {
  const int row = blockIdx.x, mat = blockIdx.y;
  const float* W = (mat == 0) ? Wq : (mat == 1) ? Wk : (mat == 2) ? Wv : Wo;
  const float* wrow = W + (long)row * Hn;
  __shared__ float red[256];
  float m = 0.0f;
  for (int c = threadIdx.x; c < Hn; c += 256) m = fmaxf(m, fabsf(wrow[c]));
  red[threadIdx.x] = m; __syncthreads();
  for (int s = 128; s > 0; s >>= 1) {
    if (threadIdx.x < s) red[threadIdx.x] = fmaxf(red[threadIdx.x], red[threadIdx.x + s]);
    __syncthreads();
  }
  float sc = fmaxf(red[0], 1e-8f) * (1.0f / 127.0f);
  if (threadIdx.x == 0) sw[mat * Hn + row] = sc;
  signed char* orow = w8 + (size_t)mat * Hn * Hn + (long)row * Hn;
  for (int c = threadIdx.x; c < Hn; c += 256)
    orow[c] = (signed char)fminf(fmaxf(roundf(wrow[c] / sc), -128.0f), 127.0f);
}

__global__ __launch_bounds__(256) void k_gemm_i8(const signed char* __restrict__ A,
                                                 const signed char* __restrict__ W,
                                                 const float* __restrict__ sw,
                                                 const float* __restrict__ bias,
                                                 const unsigned* __restrict__ sx_slot,
                                                 int Mrows, float* __restrict__ out) {
  const int lane = threadIdx.x & 31, wave = threadIdx.x >> 5;
  const int grp = lane >> 4, r = lane & 15;
  const int tileM = blockIdx.x;
  const int nBase = (blockIdx.y * 8 + wave) * 64;
  const int rowA = iclamp_hi(tileM * 16 + r, Mrows - 1);
  const signed char* arow = A + (long)rowA * Hn;
  const signed char* b0p = W + (long)(nBase +  0 + r) * Hn;
  const signed char* b1p = W + (long)(nBase + 16 + r) * Hn;
  const signed char* b2p = W + (long)(nBase + 32 + r) * Hn;
  const signed char* b3p = W + (long)(nBase + 48 + r) * Hn;
  v8i acc0 = {}, acc1 = {}, acc2 = {}, acc3 = {};
  #pragma clang loop unroll(disable)
  for (int k0 = 0; k0 < Hn; k0 += 64) {
    __builtin_prefetch(arow + k0 + 64, 0, 1);
    __builtin_prefetch(b0p + k0 + 64, 0, 1);
    v8i a  = load_a_i8(arow + k0, grp);
    v8i b0 = load_b_i8(b0p + k0, grp);
    v8i b1 = load_b_i8(b1p + k0, grp);
    v8i b2 = load_b_i8(b2p + k0, grp);
    v8i b3 = load_b_i8(b3p + k0, grp);
    acc0 = __builtin_amdgcn_wmma_i32_16x16x64_iu8(true, a, true, b0, acc0, false, false);
    acc1 = __builtin_amdgcn_wmma_i32_16x16x64_iu8(true, a, true, b1, acc1, false, false);
    acc2 = __builtin_amdgcn_wmma_i32_16x16x64_iu8(true, a, true, b2, acc2, false, false);
    acc3 = __builtin_amdgcn_wmma_i32_16x16x64_iu8(true, a, true, b3, acc3, false, false);
  }
  const float sx = fmaxf(__uint_as_float(*sx_slot), 1e-8f) * (1.0f / 127.0f);
  v8i accs[4] = {acc0, acc1, acc2, acc3};
  #pragma unroll
  for (int t = 0; t < 4; ++t) {
    const int col = nBase + t * 16 + r;
    const float bsf = sw[col] * sx;
    const float bint = roundf(bias[col] / bsf);
    #pragma unroll
    for (int j = 0; j < 8; ++j) {
      int row = tileM * 16 + j + grp * 8;
      if (row < Mrows) out[(long)row * Hn + col] = ((float)accs[t][j] + bint) * bsf;
    }
  }
}

__global__ __launch_bounds__(256) void k_requant_heads(const float* __restrict__ proj,
                                                       const unsigned* __restrict__ amax3,
                                                       signed char* __restrict__ q8,
                                                       signed char* __restrict__ k8,
                                                       _Float16* __restrict__ vt) {
  const int which = blockIdx.y;
  long idx = (long)blockIdx.x * 256 + threadIdx.x;
  if (idx >= Mn * Hn) return;
  float s = fmaxf(__uint_as_float(amax3[which]), 1e-8f) * (1.0f / 127.0f);
  float q = fminf(fmaxf(roundf(proj[(long)which * Mn * Hn + idx] / s), -128.0f), 127.0f);
  long m = idx >> 10; int col = (int)(idx & 1023);
  int b = (int)(m / Sn), sq = (int)(m % Sn);
  int h = col >> 6, d = col & 63;
  long bh = (long)b * NHn + h;
  if (which == 0)      q8[bh * Sn * HDn + (long)sq * HDn + d] = (signed char)q;
  else if (which == 1) k8[bh * Sn * HDn + (long)sq * HDn + d] = (signed char)q;
  else                 vt[bh * HDn * SPAD + (long)d * SPAD + sq] = (_Float16)(q * s);
}

__global__ __launch_bounds__(256) void k_qk(const signed char* __restrict__ q8,
                                            const signed char* __restrict__ k8,
                                            const unsigned* __restrict__ kmax_slot,
                                            float* __restrict__ attn) {
  const int lane = threadIdx.x & 31, wave = threadIdx.x >> 5;
  const int grp = lane >> 4, r = lane & 15;
  const int bh = blockIdx.z, i0 = blockIdx.x * 16;
  const int jtBase = blockIdx.y * 32 + wave * 4;
  if (jtBase * 16 >= Sn) return;
  const signed char* qb = q8 + (long)bh * Sn * HDn;
  const signed char* kb = k8 + (long)bh * Sn * HDn;
  const float scale = fmaxf(__uint_as_float(*kmax_slot), 1e-8f) * (1.0f / 127.0f) * SCALING;
  float* arow = attn + (long)bh * Sn * Sn;
  const v8i a = load_a_i8(qb + (long)iclamp_hi(i0 + r, Sn - 1) * HDn, grp);
  #pragma clang loop unroll(disable)
  for (int t = 0; t < 4; ++t) {
    const int j0 = (jtBase + t) * 16;
    if (j0 >= Sn) break;
    v8i b = load_b_i8(kb + (long)iclamp_hi(j0 + r, Sn - 1) * HDn, grp);
    v8i acc = {};
    acc = __builtin_amdgcn_wmma_i32_16x16x64_iu8(true, a, true, b, acc, false, false);
    const int col = j0 + r;
    #pragma unroll
    for (int j = 0; j < 8; ++j) {
      int row = i0 + j + grp * 8;
      if (row < Sn && col < Sn) arow[(long)row * Sn + col] = (float)acc[j] * scale;
    }
  }
}

__global__ __launch_bounds__(256) void k_softmax_a(float* __restrict__ attn,
                                                   const unsigned* __restrict__ qmax_slot,
                                                   unsigned* __restrict__ emax_slot) {
  float* x = attn + (long)blockIdx.x * Sn;
  __shared__ float red[256];
  const float sf = fmaxf(__uint_as_float(*qmax_slot), 1e-8f) * (1.0f / 127.0f);
  float m = -3.402823466e38f;
  for (int c = threadIdx.x; c < Sn; c += 256) m = fmaxf(m, x[c]);
  red[threadIdx.x] = m; __syncthreads();
  for (int s = 128; s > 0; s >>= 1) {
    if (threadIdx.x < s) red[threadIdx.x] = fmaxf(red[threadIdx.x], red[threadIdx.x + s]);
    __syncthreads();
  }
  const float rowmax = red[0]; __syncthreads();
  const float x0i = floorf(-0.6931f / sf);
  const float bi  = floorf((0.96963238f / 0.35815147f) / sf);
  const float ci  = floorf((1.0f / 0.35815147f) / (sf * sf));
  float em = 0.0f;
  for (int c = threadIdx.x; c < Sn; c += 256) {
    float xi = fmaxf(x[c] - rowmax, 30.0f * x0i);
    float qq = floorf(xi / x0i);
    float rr = xi - x0i * qq;
    float z  = rr * (rr + bi) + ci;
    float e  = fmaxf(floorf(z * exp2f(30.0f - qq)), 0.0f);
    x[c] = e;
    em = fmaxf(em, e);
  }
  red[threadIdx.x] = em; __syncthreads();
  for (int s = 128; s > 0; s >>= 1) {
    if (threadIdx.x < s) red[threadIdx.x] = fmaxf(red[threadIdx.x], red[threadIdx.x + s]);
    __syncthreads();
  }
  if (threadIdx.x == 0) atomicMax(emax_slot, __float_as_uint(red[0]));
}

__global__ __launch_bounds__(256) void k_softmax_b(const float* __restrict__ attn,
                                                   const unsigned* __restrict__ qmax_slot,
                                                   const unsigned* __restrict__ emax_slot,
                                                   _Float16* __restrict__ probs) {
  const float* x = attn + (long)blockIdx.x * Sn;
  _Float16* p = probs + (long)blockIdx.x * SPAD;
  const float sf = fmaxf(__uint_as_float(*qmax_slot), 1e-8f) * (1.0f / 127.0f);
  const float exp_sf = 0.35815147f * sf * sf * (1.0f / 1073741824.0f);
  const float s16 = fmaxf(__uint_as_float(*emax_slot) * exp_sf, 1e-8f) * (1.0f / 32767.0f);
  __shared__ float e2buf[SPAD];
  __shared__ float red[256];
  float sum = 0.0f;
  for (int c = threadIdx.x; c < SPAD; c += 256) {
    float e2 = 0.0f;
    if (c < Sn) e2 = fminf(fmaxf(roundf(x[c] * exp_sf / s16), -32768.0f), 32767.0f);
    e2buf[c] = e2; sum += e2;
  }
  red[threadIdx.x] = sum; __syncthreads();
  for (int s = 128; s > 0; s >>= 1) {
    if (threadIdx.x < s) red[threadIdx.x] += red[threadIdx.x + s];
    __syncthreads();
  }
  const float factor = floorf(4294967296.0f / red[0]);
  for (int c = threadIdx.x; c < SPAD; c += 256) {
    float pi = floorf(e2buf[c] * factor * (1.0f / 16777216.0f));
    p[c] = (_Float16)(pi * (1.0f / 256.0f));
  }
}

__global__ __launch_bounds__(256) void k_pv(const _Float16* __restrict__ probs,
                                            const _Float16* __restrict__ vt,
                                            float* __restrict__ ctx) {
  const int lane = threadIdx.x & 31, wave = threadIdx.x >> 5;
  const int grp = lane >> 4, r = lane & 15;
  const int bh = blockIdx.z, b = bh >> 4, h = bh & 15;
  const int i0 = (blockIdx.x * 8 + wave) * 16;
  if (i0 >= Sn) return;
  const _Float16* pr = probs + ((long)bh * Sn + iclamp_hi(i0 + r, Sn - 1)) * SPAD;
  const _Float16* vbase = vt + (long)bh * HDn * SPAD;
  const _Float16* v0p = vbase + (long)( 0 + r) * SPAD;
  const _Float16* v1p = vbase + (long)(16 + r) * SPAD;
  const _Float16* v2p = vbase + (long)(32 + r) * SPAD;
  const _Float16* v3p = vbase + (long)(48 + r) * SPAD;
  v8f acc0 = {}, acc1 = {}, acc2 = {}, acc3 = {};
  #pragma clang loop unroll(disable)
  for (int k0 = 0; k0 < SPAD; k0 += 32) {
    v8h lo = *(const v8h*)(pr + k0 + grp * 16);
    v8h hi = *(const v8h*)(pr + k0 + 8 + grp * 16);
    v16h a = __builtin_shufflevector(lo, hi, 0,1,2,3,4,5,6,7,8,9,10,11,12,13,14,15);
    v16h bf0 = *(const v16h*)(v0p + k0 + grp * 16);
    v16h bf1 = *(const v16h*)(v1p + k0 + grp * 16);
    v16h bf2 = *(const v16h*)(v2p + k0 + grp * 16);
    v16h bf3 = *(const v16h*)(v3p + k0 + grp * 16);
    acc0 = __builtin_amdgcn_wmma_f32_16x16x32_f16(false, a, false, bf0, (short)0, acc0, false, false);
    acc1 = __builtin_amdgcn_wmma_f32_16x16x32_f16(false, a, false, bf1, (short)0, acc1, false, false);
    acc2 = __builtin_amdgcn_wmma_f32_16x16x32_f16(false, a, false, bf2, (short)0, acc2, false, false);
    acc3 = __builtin_amdgcn_wmma_f32_16x16x32_f16(false, a, false, bf3, (short)0, acc3, false, false);
  }
  v8f accs[4] = {acc0, acc1, acc2, acc3};
  #pragma unroll
  for (int t = 0; t < 4; ++t) {
    const int d = t * 16 + r;
    #pragma unroll
    for (int j = 0; j < 8; ++j) {
      int i = i0 + j + grp * 8;
      if (i < Sn) ctx[((long)b * Sn + i) * Hn + h * HDn + d] = accs[t][j];
    }
  }
}

extern "C" void kernel_launch(void* const* d_in, const int* in_sizes, int n_in,
                              void* d_out, int out_size, void* d_ws, size_t ws_size,
                              hipStream_t stream) {
  (void)in_sizes; (void)n_in; (void)out_size; (void)ws_size;
  const float* hid = (const float*)d_in[0];
  const float* Wq  = (const float*)d_in[1]; const float* bq = (const float*)d_in[2];
  const float* Wk  = (const float*)d_in[3]; const float* bk = (const float*)d_in[4];
  const float* Wv  = (const float*)d_in[5]; const float* bv = (const float*)d_in[6];
  const float* Wo  = (const float*)d_in[7]; const float* bo = (const float*)d_in[8];

  char* ws = (char*)d_ws;
  unsigned*    scal = (unsigned*)(ws + OFF_SCAL);
  signed char* x8   = (signed char*)(ws + OFF_X8);
  signed char* w8   = (signed char*)(ws + OFF_W8);
  float*       sw   = (float*)(ws + OFF_SW);
  float*       proj = (float*)(ws + OFF_PROJ);
  signed char* q8   = (signed char*)(ws + OFF_Q8);
  signed char* k8   = (signed char*)(ws + OFF_K8);
  _Float16*    vt   = (_Float16*)(ws + OFF_VT);
  float*       attn = (float*)(ws + OFF_ATTN);
  _Float16*    prb  = (_Float16*)(ws + OFF_PRB);
  float*       out  = (float*)d_out;
  float*       ctx  = proj;
  signed char* ctx8 = x8;

  const long nXH = Mn * Hn;
  const int  gElem = (int)((nXH + 255) / 256);
  const long nVT = (long)BH * HDn * SPAD;
  const int  mTiles = (int)((Mn + 15) / 16);
  const int  sTiles = (Sn + 15) / 16;
  const int  nRows = BH * Sn;

  k_init<<<4096, 256, 0, stream>>>(scal, vt, nVT);
  k_absmax<<<2048, 256, 0, stream>>>(hid, nXH, scal + 0);
  k_quant8<<<gElem, 256, 0, stream>>>(hid, scal + 0, x8, nXH);
  k_quantw<<<dim3(Hn, 4), 256, 0, stream>>>(Wq, Wk, Wv, Wo, w8, sw);
  k_gemm_i8<<<dim3(mTiles, 2), 256, 0, stream>>>(x8, w8 + 0ull*Hn*Hn, sw + 0*Hn, bq, scal + 0, (int)Mn, proj + 0*nXH);
  k_gemm_i8<<<dim3(mTiles, 2), 256, 0, stream>>>(x8, w8 + 1ull*Hn*Hn, sw + 1*Hn, bk, scal + 0, (int)Mn, proj + 1*nXH);
  k_gemm_i8<<<dim3(mTiles, 2), 256, 0, stream>>>(x8, w8 + 2ull*Hn*Hn, sw + 2*Hn, bv, scal + 0, (int)Mn, proj + 2*nXH);
  k_absmax<<<2048, 256, 0, stream>>>(proj + 0*nXH, nXH, scal + 1);
  k_absmax<<<2048, 256, 0, stream>>>(proj + 1*nXH, nXH, scal + 2);
  k_absmax<<<2048, 256, 0, stream>>>(proj + 2*nXH, nXH, scal + 3);
  k_requant_heads<<<dim3(gElem, 3), 256, 0, stream>>>(proj, scal + 1, q8, k8, vt);
  k_qk<<<dim3(sTiles, 2, BH), 256, 0, stream>>>(q8, k8, scal + 2, attn);
  k_softmax_a<<<nRows, 256, 0, stream>>>(attn, scal + 1, scal + 4);
  k_softmax_b<<<nRows, 256, 0, stream>>>(attn, scal + 1, scal + 4, prb);
  k_pv<<<dim3((sTiles + 7) / 8, 1, BH), 256, 0, stream>>>(prb, vt, ctx);
  k_absmax<<<2048, 256, 0, stream>>>(ctx, nXH, scal + 5);
  k_quant8<<<gElem, 256, 0, stream>>>(ctx, scal + 5, ctx8, nXH);
  k_gemm_i8<<<dim3(mTiles, 2), 256, 0, stream>>>(ctx8, w8 + 3ull*Hn*Hn, sw + 3*Hn, bo, scal + 5, (int)Mn, out);
}
